// MultiHeadAttention_80891414053295
// MI455X (gfx1250) — compile-verified, matrix operand order checked by reading, not run on hardware
//
#include <hip/hip_runtime.h>
#include <hip/hip_bf16.h>
#include <math.h>

typedef __bf16 bf16;
typedef __attribute__((ext_vector_type(16))) bf16  v16bf;
typedef __attribute__((ext_vector_type(8)))  bf16  v8bf;
typedef __attribute__((ext_vector_type(8)))  float v8f;
typedef __attribute__((ext_vector_type(4)))  int   v4i;

#define B_   4
#define L_   1536
#define DM_  512
#define H_   8
#define C_   512
#define BL_  (B_*L_)
#define P_   4
#define LK_  171          // ceil(1536/9) subsampled keys
#define SCN_ 192          // LK_ padded to a multiple of 32 for WMMA-K

#define AS1 __attribute__((address_space(1)))
#define AS3 __attribute__((address_space(3)))

#if __has_builtin(__builtin_amdgcn_global_load_async_to_lds_b128) && \
    __has_builtin(__builtin_amdgcn_s_wait_asynccnt)
#define USE_ASYNC 1
#else
#define USE_ASYNC 0
#endif

#define CAT16(lo,hi) __builtin_shufflevector(lo,hi,0,1,2,3,4,5,6,7,8,9,10,11,12,13,14,15)

__device__ __forceinline__ v8bf zero8() {
    v8bf r;
#pragma unroll
    for (int i = 0; i < 8; ++i) r[i] = (bf16)0.0f;
    return r;
}

__device__ __forceinline__ v8f wmma_bf16(v16bf a, v16bf b, v8f c) {
    // (neg_a, A, neg_b, B, c_mod, C, reuse_a, reuse_b)
    return __builtin_amdgcn_wmma_f32_16x16x32_bf16(false, a, false, b, (short)0, c, false, false);
}

__device__ __forceinline__ void async_cp16(const bf16* g, bf16* l) {
#if USE_ASYNC
    __builtin_amdgcn_global_load_async_to_lds_b128((AS1 v4i*)g, (AS3 v4i*)l, 0, 0);
#else
    (void)g; (void)l;
#endif
}

__device__ __forceinline__ void wait_async0() {
#if USE_ASYNC
    __builtin_amdgcn_s_wait_asynccnt(0);
#endif
}

// ---------------------------------------------------------------- converts
__global__ void cvt_kernel(bf16* __restrict__ dst, const float* __restrict__ src, long n) {
    long i = (long)blockIdx.x * blockDim.x + threadIdx.x;
    if (i < n) dst[i] = (bf16)src[i];
}

__global__ void zeroz_kernel(unsigned int* __restrict__ p) {
    p[threadIdx.x] = 0u;   // 64 dwords = 256B guard buffer
}

// conv weight (C,C,f) -> bf16 [tau][o][i]  (each tau slice is a row-major BT)
__global__ void convw_kernel(bf16* __restrict__ dst, const float* __restrict__ src, int f) {
    long i = (long)blockIdx.x * blockDim.x + threadIdx.x;
    long n = (long)C_ * C_ * f;
    if (i >= n) return;
    int tau = (int)(i % f);
    long oi = i / f;                 // o*C + i
    dst[(long)tau * C_ * C_ + oi] = (bf16)src[i];
}

// ---------------------------------------------------------------- generic WMMA GEMM
// out[row, col] = sum_tau sum_k A[row + shiftBase + tau, k] * BT[col, k]   (BT k-major)
// rows shifted outside their Lseg segment read as zero (conv padding) via zsrc redirect.
// Block tile 256x64, 8 waves (wave tile 64x32 = 8 WMMA / k-step),
// double-buffered LDS fed by GLOBAL_LOAD_ASYNC_TO_LDS_B128 when available.
#define BM_ 256
#define BN_ 64
#define LDSA_ 40   // padded row stride in halves (80B: conflict-free, 16B aligned)

__global__ __launch_bounds__(256) void gemm_kernel(
    const bf16* __restrict__ A, const bf16* __restrict__ BT,
    float* __restrict__ outF, bf16* __restrict__ outH,
    const bf16* __restrict__ zsrc,
    int M, int N, int K, int lda, int ldbt, int ldc,
    long aStride, long btStride, long cStride,
    int tauCount, long tauStride, int shiftBase, int Lseg)
{
    const int z = blockIdx.z;
    A  += (long)z * aStride;
    BT += (long)z * btStride;
    if (outF) outF += (long)z * cStride;
    if (outH) outH += (long)z * cStride;

    const int tid  = threadIdx.x;
    const int lane = tid & 31;
    const int wid  = tid >> 5;
    const int wm = wid & 3, wn = wid >> 2;           // 4x2 wave grid -> 256x64 tile
    const int mBase = blockIdx.y * BM_;
    const int nBase = blockIdx.x * BN_;
    const int l15 = lane & 15, sel = lane >> 4;

#if USE_ASYNC
    __shared__ bf16 As[2 * BM_ * LDSA_];
    __shared__ bf16 Bs[2 * BN_ * LDSA_];
#else
    __shared__ bf16 As[BM_ * LDSA_];
    __shared__ bf16 Bs[BN_ * LDSA_];
#endif

    v8f acc[4][2];
#pragma unroll
    for (int i = 0; i < 4; ++i)
#pragma unroll
        for (int j = 0; j < 2; ++j)
#pragma unroll
            for (int e = 0; e < 8; ++e) acc[i][j][e] = 0.0f;

    const int ksteps = K >> 5;
    const int S = tauCount * ksteps;

#if USE_ASYNC
    // stage s -> issue async copies of A(256x32) + B(64x32) into buffer s&1
    auto issue = [&](int s) {
        const int tau = s / ksteps;
        const int k0  = (s - tau * ksteps) << 5;
        const int shift = shiftBase + tau;
        const bf16* BTt = BT + (long)tau * tauStride;
        bf16* AsB = As + (s & 1) * (BM_ * LDSA_);
        bf16* BsB = Bs + (s & 1) * (BN_ * LDSA_);
#pragma unroll
        for (int j = 0; j < 4; ++j) {                 // A: 1024 16B chunks / 256 thr
            int idx = tid + j * 256;
            int row = idx >> 2, sub = idx & 3;
            int outRow = mBase + row;
            int gRow = outRow + shift;
            int seg = outRow / Lseg;
            bool v = (outRow < M) && (gRow >= seg * Lseg) && (gRow < (seg + 1) * Lseg);
            const bf16* src = v ? (A + (long)gRow * lda + k0 + sub * 8) : zsrc;
            async_cp16(src, AsB + row * LDSA_ + sub * 8);
        }
        {                                              // B: 256 16B chunks / 256 thr
            int row = tid >> 2, sub = tid & 3;
            int col = nBase + row;
            const bf16* src = (col < N) ? (BTt + (long)col * ldbt + k0 + sub * 8) : zsrc;
            async_cp16(src, BsB + row * LDSA_ + sub * 8);
        }
    };

    issue(0);
    for (int s = 0; s < S; ++s) {
        wait_async0();
        __syncthreads();
        if (s + 1 < S) issue(s + 1);
        const bf16* AsB = As + (s & 1) * (BM_ * LDSA_);
        const bf16* BsB = Bs + (s & 1) * (BN_ * LDSA_);
        v16bf af[4], bfv[2];
#pragma unroll
        for (int ti = 0; ti < 4; ++ti) {
            int r = wm * 64 + ti * 16 + l15;
            v8bf lo = *(const v8bf*)(AsB + r * LDSA_ + sel * 8);
            v8bf hi = *(const v8bf*)(AsB + r * LDSA_ + 16 + sel * 8);
            af[ti] = CAT16(lo, hi);
        }
#pragma unroll
        for (int tj = 0; tj < 2; ++tj) {
            int r = wn * 32 + tj * 16 + l15;
            v8bf lo = *(const v8bf*)(BsB + r * LDSA_ + sel * 8);
            v8bf hi = *(const v8bf*)(BsB + r * LDSA_ + 16 + sel * 8);
            bfv[tj] = CAT16(lo, hi);
        }
#pragma unroll
        for (int ti = 0; ti < 4; ++ti)
#pragma unroll
            for (int tj = 0; tj < 2; ++tj)
                acc[ti][tj] = wmma_bf16(af[ti], bfv[tj], acc[ti][tj]);
    }
#else
    // fallback: register-staged single-buffer pipeline
    for (int s = 0; s < S; ++s) {
        const int tau = s / ksteps;
        const int k0  = (s - tau * ksteps) << 5;
        const int shift = shiftBase + tau;
        const bf16* BTt = BT + (long)tau * tauStride;
        v8bf aR[4];
#pragma unroll
        for (int j = 0; j < 4; ++j) {
            int idx = tid + j * 256;
            int row = idx >> 2, sub = idx & 3;
            int outRow = mBase + row;
            int gRow = outRow + shift;
            int seg = outRow / Lseg;
            bool v = (outRow < M) && (gRow >= seg * Lseg) && (gRow < (seg + 1) * Lseg);
            aR[j] = v ? *(const v8bf*)(A + (long)gRow * lda + k0 + sub * 8) : zero8();
        }
        v8bf bR;
        {
            int row = tid >> 2, sub = tid & 3;
            int col = nBase + row;
            bR = (col < N) ? *(const v8bf*)(BTt + (long)col * ldbt + k0 + sub * 8) : zero8();
        }
        __syncthreads();
#pragma unroll
        for (int j = 0; j < 4; ++j) {
            int idx = tid + j * 256;
            int row = idx >> 2, sub = idx & 3;
            *(v8bf*)(As + row * LDSA_ + sub * 8) = aR[j];
        }
        *(v8bf*)(Bs + (tid >> 2) * LDSA_ + (tid & 3) * 8) = bR;
        __syncthreads();
        v16bf af[4], bfv[2];
#pragma unroll
        for (int ti = 0; ti < 4; ++ti) {
            int r = wm * 64 + ti * 16 + l15;
            v8bf lo = *(const v8bf*)(As + r * LDSA_ + sel * 8);
            v8bf hi = *(const v8bf*)(As + r * LDSA_ + 16 + sel * 8);
            af[ti] = CAT16(lo, hi);
        }
#pragma unroll
        for (int tj = 0; tj < 2; ++tj) {
            int r = wn * 32 + tj * 16 + l15;
            v8bf lo = *(const v8bf*)(Bs + r * LDSA_ + sel * 8);
            v8bf hi = *(const v8bf*)(Bs + r * LDSA_ + 16 + sel * 8);
            bfv[tj] = CAT16(lo, hi);
        }
#pragma unroll
        for (int ti = 0; ti < 4; ++ti)
#pragma unroll
            for (int tj = 0; tj < 2; ++tj)
                acc[ti][tj] = wmma_bf16(af[ti], bfv[tj], acc[ti][tj]);
    }
#endif

#pragma unroll
    for (int ti = 0; ti < 4; ++ti)
#pragma unroll
        for (int tj = 0; tj < 2; ++tj)
#pragma unroll
            for (int j = 0; j < 8; ++j) {
                int row = mBase + wm * 64 + ti * 16 + j + sel * 8;
                int col = nBase + wn * 32 + tj * 16 + l15;
                if (row < M && col < N) {
                    float v = acc[ti][tj][j];
                    if (outF) outF[(long)row * ldc + col] = v;
                    if (outH) outH[(long)row * ldc + col] = (bf16)v;
                }
            }
}

// ---------------------------------------------------------------- BN stats + ELU
__global__ void bnstats_kernel(const float* __restrict__ Y, float* __restrict__ mu, float* __restrict__ rs) {
    int c = blockIdx.x, tid = threadIdx.x;
    float s = 0.f, s2 = 0.f;
    for (int t = tid; t < BL_; t += 256) {
        float v = Y[(long)t * C_ + c];
        s += v; s2 += v * v;
    }
    __shared__ float sh[256], sh2[256];
    sh[tid] = s; sh2[tid] = s2; __syncthreads();
    for (int o = 128; o > 0; o >>= 1) {
        if (tid < o) { sh[tid] += sh[tid + o]; sh2[tid] += sh2[tid + o]; }
        __syncthreads();
    }
    if (tid == 0) {
        float m = sh[0] / (float)BL_;
        float var = sh2[0] / (float)BL_ - m * m;
        mu[c] = m;
        rs[c] = rsqrtf(var + 1e-5f);
    }
}

__global__ void bnelu_kernel(const float* __restrict__ Y, const float* __restrict__ mu,
                             const float* __restrict__ rs, bf16* __restrict__ O) {
    long i = (long)blockIdx.x * blockDim.x + threadIdx.x;
    if (i >= (long)BL_ * C_) return;
    int c = (int)(i % C_);
    float y = (Y[i] - mu[c]) * rs[c];
    O[i] = (bf16)(y > 0.f ? y : (__expf(y) - 1.f));
}

// ---------------------------------------------------------------- scores + softmax + max over P + exp
struct ScoreArgs { const bf16* q[P_]; const bf16* k[P_]; };

__global__ __launch_bounds__(128) void scores_kernel(ScoreArgs sa, bf16* __restrict__ E, float* __restrict__ SE) {
    const int tid = threadIdx.x;
    const int lane = tid & 31;
    const int p = tid >> 5;              // wave = filter index
    const int q0 = blockIdx.x * 16;
    const int h = blockIdx.y;
    const int b = blockIdx.z;
    const int l15 = lane & 15, sel = lane >> 4;

    __shared__ float s[P_][16][SCN_];

    v8f acc[12];
#pragma unroll
    for (int n = 0; n < 12; ++n)
#pragma unroll
        for (int j = 0; j < 8; ++j) acc[n][j] = 0.0f;

    const bf16* Qbase = sa.q[p] + ((long)(b * L_ + q0 + l15)) * C_ + h * 64;
    const bf16* Kside = sa.k[p];

#pragma unroll
    for (int ks = 0; ks < 64; ks += 32) {
        v8bf alo = *(const v8bf*)(Qbase + ks + sel * 8);
        v8bf ahi = *(const v8bf*)(Qbase + ks + 16 + sel * 8);
        v16bf a = CAT16(alo, ahi);
#pragma unroll
        for (int n = 0; n < 12; ++n) {
            int kp = n * 16 + l15;
            v8bf blo = zero8(), bhi = zero8();
            if (kp < LK_) {
                const bf16* kr = Kside + ((long)(b * L_ + 9 * kp)) * C_ + h * 64 + ks;
                blo = *(const v8bf*)(kr + sel * 8);
                bhi = *(const v8bf*)(kr + 16 + sel * 8);
            }
            v16bf bb = CAT16(blo, bhi);
            acc[n] = wmma_bf16(a, bb, acc[n]);
        }
    }

#pragma unroll
    for (int n = 0; n < 12; ++n)
#pragma unroll
        for (int j = 0; j < 8; ++j) {
            int row = j + (sel << 3);
            s[p][row][n * 16 + l15] = acc[n][j] * 0.125f;   // 1/sqrt(DK)
        }
    __syncthreads();

    // per (p,row) softmax over the 171 valid keys (serial -> deterministic)
    if (tid < 64) {
        int pp = tid >> 4, row = tid & 15;
        float mx = -1e30f;
        for (int c = 0; c < LK_; ++c) mx = fmaxf(mx, s[pp][row][c]);
        float sm = 0.f;
        for (int c = 0; c < LK_; ++c) { float e = __expf(s[pp][row][c] - mx); s[pp][row][c] = e; sm += e; }
        float inv = 1.f / sm;
        for (int c = 0; c < LK_; ++c) s[pp][row][c] *= inv;
    }
    __syncthreads();

    // max over P, exp, store E (zero-padded), stash exp into s[0] for row sums
    for (int idx = tid; idx < 16 * SCN_; idx += 128) {
        int row = idx / SCN_, c = idx % SCN_;
        long eoff = (((long)(b * H_ + h)) * L_ + q0 + row) * SCN_ + c;
        if (c < LK_) {
            float a4 = fmaxf(fmaxf(s[0][row][c], s[1][row][c]), fmaxf(s[2][row][c], s[3][row][c]));
            float e = __expf(a4);
            s[0][row][c] = e;
            E[eoff] = (bf16)e;
        } else {
            E[eoff] = (bf16)0.0f;
        }
    }
    __syncthreads();
    if (tid < 16) {
        float sm = 0.f;
        for (int c = 0; c < LK_; ++c) sm += s[0][tid][c];
        SE[((long)(b * H_ + h)) * L_ + q0 + tid] = sm;
    }
}

// ---------------------------------------------------------------- V helpers
// VsubT[b,h,d,c] = VS[b, 9c, h*64+d] (zero pad c>=LK_)
__global__ void vsub_kernel(bf16* __restrict__ VT, const bf16* __restrict__ VS) {
    long i = (long)blockIdx.x * blockDim.x + threadIdx.x;
    long n = (long)B_ * H_ * 64 * SCN_;
    if (i >= n) return;
    int c = (int)(i % SCN_);
    int d = (int)((i / SCN_) % 64);
    int h = (int)((i / ((long)SCN_ * 64)) % H_);
    int b = (int)(i / ((long)SCN_ * 64 * H_));
    bf16 v = (bf16)0.0f;
    if (c < LK_) v = VS[((long)(b * L_ + 9 * c)) * C_ + h * 64 + d];
    VT[i] = v;
}

__global__ void sumv_kernel(const bf16* __restrict__ VS, float* __restrict__ sAll, float* __restrict__ sSub) {
    int bh = blockIdx.x;
    int b = bh / H_, h = bh % H_;
    int tid = threadIdx.x;
    int d = tid & 63, part = tid >> 6;                 // 4 partials per d
    float sa = 0.f, ss = 0.f;
    for (int i = 0; i < 384; ++i) {
        int t = part * 384 + i;
        sa += (float)VS[((long)(b * L_ + t)) * C_ + h * 64 + d];
    }
    for (int i = 0; i < 43; ++i) {
        int c = part * 43 + i;
        if (c < LK_) ss += (float)VS[((long)(b * L_ + 9 * c)) * C_ + h * 64 + d];
    }
    __shared__ float shA[256], shS[256];
    shA[tid] = sa; shS[tid] = ss; __syncthreads();
    if (part == 0) {
        float a = shA[d] + shA[64 + d] + shA[128 + d] + shA[192 + d];
        float s = shS[d] + shS[64 + d] + shS[128 + d] + shS[192 + d];
        sAll[(long)bh * 64 + d] = a;
        sSub[(long)bh * 64 + d] = s;
    }
}

// ctx[b,t,h*64+d] = (sumAll - sumSub + EV) / (SE + (L-LK))
__global__ void combine_kernel(bf16* __restrict__ CTX, const float* __restrict__ EV,
                               const float* __restrict__ sAll, const float* __restrict__ sSub,
                               const float* __restrict__ SE) {
    long i = (long)blockIdx.x * blockDim.x + threadIdx.x;
    if (i >= (long)BL_ * C_) return;
    int c = (int)(i % C_);
    long bt = i / C_;
    int t = (int)(bt % L_);
    int b = (int)(bt / L_);
    int h = c >> 6, d = c & 63;
    long bh = (long)b * H_ + h;
    float num = sAll[bh * 64 + d] - sSub[bh * 64 + d] + EV[(bh * L_ + t) * 64 + d];
    float den = SE[bh * L_ + t] + (float)(L_ - LK_);
    CTX[i] = (bf16)(num / den);
}

// ---------------------------------------------------------------- host
static inline size_t aln(size_t x) { return (x + 255) & ~(size_t)255; }

extern "C" void kernel_launch(void* const* d_in, const int* in_sizes, int n_in,
                              void* d_out, int out_size, void* d_ws, size_t ws_size,
                              hipStream_t stream) {
    (void)in_sizes; (void)n_in; (void)out_size; (void)ws_size;
    const float* Q   = (const float*)d_in[0];
    const float* K   = (const float*)d_in[1];
    const float* V   = (const float*)d_in[2];
    const float* WQ  = (const float*)d_in[4];
    const float* WK  = (const float*)d_in[5];
    const float* WV  = (const float*)d_in[6];
    const float* Wfc = (const float*)d_in[7];
    const float* cqw[P_] = {(const float*)d_in[8],  (const float*)d_in[9],
                            (const float*)d_in[10], (const float*)d_in[11]};
    const float* ckw[P_] = {(const float*)d_in[12], (const float*)d_in[13],
                            (const float*)d_in[14], (const float*)d_in[15]};
    const int filt[P_] = {1, 3, 6, 9};

    char* cur = (char*)d_ws;
    auto take = [&](size_t bytes) { char* p = cur; cur += aln(bytes); return p; };

    bf16* zbuf = (bf16*)take(256);
    bf16* Qb  = (bf16*)take((size_t)BL_ * DM_ * 2);
    bf16* Kb  = (bf16*)take((size_t)BL_ * DM_ * 2);
    bf16* Vb  = (bf16*)take((size_t)BL_ * DM_ * 2);
    bf16* WqB = (bf16*)take((size_t)C_ * DM_ * 2);
    bf16* WkB = (bf16*)take((size_t)C_ * DM_ * 2);
    bf16* WvB = (bf16*)take((size_t)C_ * DM_ * 2);
    bf16* WfcB= (bf16*)take((size_t)DM_ * C_ * 2);
    bf16* CWq[P_]; bf16* CWk[P_];
    for (int i = 0; i < P_; ++i) CWq[i] = (bf16*)take((size_t)filt[i] * C_ * C_ * 2);
    for (int i = 0; i < P_; ++i) CWk[i] = (bf16*)take((size_t)filt[i] * C_ * C_ * 2);
    bf16* QS = (bf16*)take((size_t)BL_ * C_ * 2);
    bf16* KS = (bf16*)take((size_t)BL_ * C_ * 2);
    bf16* VS = (bf16*)take((size_t)BL_ * C_ * 2);
    float* Yacc = (float*)take((size_t)BL_ * C_ * 4);
    bf16* Ql[P_]; bf16* Kl[P_];
    for (int i = 0; i < P_; ++i) Ql[i] = (bf16*)take((size_t)BL_ * C_ * 2);
    for (int i = 0; i < P_; ++i) Kl[i] = (bf16*)take((size_t)BL_ * C_ * 2);
    float* mu = (float*)take((size_t)C_ * 4);
    float* rs = (float*)take((size_t)C_ * 4);
    bf16* E  = (bf16*)take((size_t)B_ * H_ * L_ * SCN_ * 2);
    float* SE = (float*)take((size_t)B_ * H_ * L_ * 4);
    bf16* VT = (bf16*)take((size_t)B_ * H_ * 64 * SCN_ * 2);
    float* sAll = (float*)take((size_t)B_ * H_ * 64 * 4);
    float* sSub = (float*)take((size_t)B_ * H_ * 64 * 4);
    float* EV = (float*)take((size_t)B_ * H_ * L_ * 64 * 4);
    bf16* CTX = (bf16*)take((size_t)BL_ * C_ * 2);

    auto cvt = [&](bf16* d, const float* s, long n) {
        cvt_kernel<<<dim3((unsigned)((n + 255) / 256)), 256, 0, stream>>>(d, s, n);
    };
    auto gemm = [&](const bf16* A, const bf16* BT, float* oF, bf16* oH,
                    int M, int N, int Kk, int lda, int ldbt, int ldc,
                    int batches, long aS, long bS, long cS,
                    int tauC, long tauS, int shiftBase, int Lseg) {
        dim3 g((unsigned)(N / BN_), (unsigned)((M + BM_ - 1) / BM_), (unsigned)batches);
        gemm_kernel<<<g, 256, 0, stream>>>(A, BT, oF, oH, zbuf, M, N, Kk, lda, ldbt, ldc,
                                           aS, bS, cS, tauC, tauS, shiftBase, Lseg);
    };

    // 0) zero guard buffer for async padding reads
    zeroz_kernel<<<dim3(1), 64, 0, stream>>>((unsigned int*)zbuf);

    // 1) convert activations + weights to bf16
    cvt(Qb, Q, (long)BL_ * DM_);
    cvt(Kb, K, (long)BL_ * DM_);
    cvt(Vb, V, (long)BL_ * DM_);
    cvt(WqB, WQ, (long)C_ * DM_);
    cvt(WkB, WK, (long)C_ * DM_);
    cvt(WvB, WV, (long)C_ * DM_);
    cvt(WfcB, Wfc, (long)DM_ * C_);
    for (int i = 0; i < P_; ++i) {
        long n = (long)C_ * C_ * filt[i];
        convw_kernel<<<dim3((unsigned)((n + 255) / 256)), 256, 0, stream>>>(CWq[i], cqw[i], filt[i]);
        convw_kernel<<<dim3((unsigned)((n + 255) / 256)), 256, 0, stream>>>(CWk[i], ckw[i], filt[i]);
    }

    // 2) projections (bf16 out)
    gemm(Qb, WqB, nullptr, QS, BL_, C_, DM_, DM_, DM_, C_, 1, 0, 0, 0, 1, 0, 0, L_);
    gemm(Kb, WkB, nullptr, KS, BL_, C_, DM_, DM_, DM_, C_, 1, 0, 0, 0, 1, 0, 0, L_);
    gemm(Vb, WvB, nullptr, VS, BL_, C_, DM_, DM_, DM_, C_, 1, 0, 0, 0, 1, 0, 0, L_);

    // 3) convs as tau-accumulated GEMMs + BN + ELU
    for (int side = 0; side < 2; ++side) {
        const bf16* X = side ? KS : QS;
        for (int i = 0; i < P_; ++i) {
            int f = filt[i];
            const bf16* W = side ? CWk[i] : CWq[i];
            bf16* O = side ? Kl[i] : Ql[i];
            gemm(X, W, Yacc, nullptr, BL_, C_, C_, C_, C_, C_,
                 1, 0, 0, 0, f, (long)C_ * C_, -(f / 2), L_);
            bnstats_kernel<<<dim3(C_), 256, 0, stream>>>(Yacc, mu, rs);
            bnelu_kernel<<<dim3((unsigned)(((long)BL_ * C_ + 255) / 256)), 256, 0, stream>>>(Yacc, mu, rs, O);
        }
    }

    // 4) scores + softmax + max over P + exp  -> E (bf16, zero-padded), SE
    ScoreArgs sa;
    for (int i = 0; i < P_; ++i) { sa.q[i] = Ql[i]; sa.k[i] = Kl[i]; }
    scores_kernel<<<dim3(L_ / 16, H_, B_), 128, 0, stream>>>(sa, E, SE);

    // 5) V helpers + EV GEMM + combine
    {
        long n = (long)B_ * H_ * 64 * SCN_;
        vsub_kernel<<<dim3((unsigned)((n + 255) / 256)), 256, 0, stream>>>(VT, VS);
    }
    sumv_kernel<<<dim3(B_ * H_), 256, 0, stream>>>(VS, sAll, sSub);
    gemm(E, VT, EV, nullptr, L_, 64, SCN_, SCN_, SCN_, 64,
         B_ * H_, (long)L_ * SCN_, (long)64 * SCN_, (long)L_ * 64, 1, 0, 0, L_);
    combine_kernel<<<dim3((unsigned)(((long)BL_ * C_ + 255) / 256)), 256, 0, stream>>>(CTX, EV, sAll, sSub, SE);

    // 6) output projection -> d_out (f32)
    gemm(CTX, WfcB, (float*)d_out, nullptr, BL_, DM_, C_, C_, C_, DM_, 1, 0, 0, 0, 1, 0, 0, L_);
}
